// LocalAttention_21062519620142
// MI455X (gfx1250) — matrix operand order repaired and checked by reading, not compiled, not run on hardware
//
#include <hip/hip_runtime.h>

// ---------------------------------------------------------------------------
// Windowed multi-head attention for MI455X (gfx1250), wave32 + WMMA f16.
//   B=8, C=512, H=W=128, WS=8 -> N=2048 windows, L=64 tokens, NH=4, hd=128
// One workgroup (256 threads = 8 wave32) processes one window fully fused:
//   x gather -> QKV GEMM -> per-head attention -> out-proj GEMM -> scatter.
// All GEMMs use v_wmma_f32_16x16x32_f16 (f16 in, f32 accumulate).
// ---------------------------------------------------------------------------

typedef _Float16 h8  __attribute__((ext_vector_type(8)));
typedef _Float16 h16 __attribute__((ext_vector_type(16)));
typedef float    f8v __attribute__((ext_vector_type(8)));

#define C_   512
#define HW_  128
#define L_   64
#define NH_  4
#define HD_  128

// padded LDS row strides (f16 elements, multiples of 8 for 16B alignment)
#define LDX  520   // X tile / ctx tile rows (512 + 8)
#define LDQ  136   // Q/K head-slice rows (128 + 8)
#define LDV  72    // V-transposed rows (64 + 8)
#define LDSS 65    // scores rows (f32)
#define LDP  72    // softmax P rows (64 + 8)

// A-matrix fragment, 16x32 f16, row-major source with leading dim `ld`.
// Per ISA layout: lane<16 -> K {0..7, 16..23}; lane>=16 -> K {8..15, 24..31}.
__device__ __forceinline__ h16 ldfragA(const _Float16* p, int ld, int m0, int k0, int lane) {
  const _Float16* r = p + (size_t)(m0 + (lane & 15)) * ld + k0 + ((lane >> 4) << 3);
  h8 lo = *(const h8*)r;
  h8 hi = *(const h8*)(r + 16);
  return __builtin_shufflevector(lo, hi, 0,1,2,3,4,5,6,7,8,9,10,11,12,13,14,15);
}

// B-matrix fragment from Bt storage (row n = output column, K contiguous).
// Per ISA layout: lane<16 -> K {0..7, 16..23}; lane>=16 -> K {8..15, 24..31}, col = lane&15.
__device__ __forceinline__ h16 ldfragB(const _Float16* p, int ld, int n0, int k0, int lane) {
  const _Float16* r = p + (size_t)(n0 + (lane & 15)) * ld + k0 + ((lane >> 4) << 3);
  h8 lo = *(const h8*)r;
  h8 hi = *(const h8*)(r + 16);
  return __builtin_shufflevector(lo, hi, 0,1,2,3,4,5,6,7,8,9,10,11,12,13,14,15);
}

__device__ __forceinline__ f8v wmma_f16(h16 a, h16 b, f8v c) {
  return __builtin_amdgcn_wmma_f32_16x16x32_f16(false, a, false, b, (short)0, c, false, false);
}

// ---------------------------------------------------------------------------
// Prep: convert w_in (1536x512) and w_out (512x512) f32 -> f16 into d_ws.
// ---------------------------------------------------------------------------
__global__ void cvt_weights_kernel(const float* __restrict__ w_in,
                                   const float* __restrict__ w_out,
                                   _Float16* __restrict__ w16,
                                   int n_in, int n_total) {
  int i = blockIdx.x * blockDim.x + threadIdx.x;
  if (i < n_in)            w16[i] = (_Float16)w_in[i];
  else if (i < n_total)    w16[i] = (_Float16)w_out[i - n_in];
}

// ---------------------------------------------------------------------------
// Fused per-window attention kernel. grid = 2048 blocks of 256 threads.
// ---------------------------------------------------------------------------
__global__ __launch_bounds__(256)
void win_attn_kernel(const float* __restrict__ x,
                     const _Float16* __restrict__ w_in16,   // (1536, 512)
                     const float* __restrict__ b_in,        // (1536,)
                     const _Float16* __restrict__ w_out16,  // (512, 512)
                     const float* __restrict__ b_out,       // (512,)
                     float* __restrict__ out) {
  __shared__ __align__(16) _Float16 Xs[L_ * LDX];    // window x tile, f16
  __shared__ __align__(16) _Float16 Cx[L_ * LDX];    // ctx (all heads), f16
  __shared__ __align__(16) _Float16 Qs[L_ * LDQ];    // Q head slice
  __shared__ __align__(16) _Float16 Ks[L_ * LDQ];    // K head slice
  __shared__ __align__(16) _Float16 Vt[HD_ * LDV];   // V head slice, transposed (d, t)
  __shared__ __align__(16) float    Sb[L_ * LDSS];   // scores f32
  __shared__ __align__(16) _Float16 Pb[L_ * LDP];    // softmax probs f16

  const int tid  = threadIdx.x;
  const int lane = tid & 31;
  const int wave = tid >> 5;
  const int win  = blockIdx.x;          // [0, 2048)
  const int bb   = win >> 8;            // batch
  const int hwi  = (win >> 4) & 15;     // window row index
  const int wwi  = win & 15;            // window col index
  const int row0 = hwi * 8;             // pixel row base
  const int col0 = wwi * 8;             // pixel col base
  const size_t xbase = (size_t)bb * C_ * HW_ * HW_;

  // ---- Stage 1: gather window (64 tokens x 512 channels) into LDS as f16 ----
  for (int idx = tid; idx < L_ * C_; idx += 256) {
    int c = idx >> 6;           // channel
    int l = idx & 63;           // token in window
    int r = l >> 3, s = l & 7;
    float v = x[xbase + ((size_t)c * HW_ + (row0 + r)) * HW_ + (col0 + s)];
    Xs[l * LDX + c] = (_Float16)v;
  }
  __syncthreads();

  const float scale = 0.08838834764831845f;   // 1/sqrt(128)

  for (int h = 0; h < NH_; ++h) {
    // ---- Stage 2: QKV slices for head h:  (64 x 384) = Xs(64x512) @ Win^T ----
    // 96 output tiles: seg {q,k,v} x 8 n-tiles x 4 m-tiles, K = 512 (16 steps)
    for (int tile = wave; tile < 96; tile += 8) {
      int mt  = tile & 3;
      int nn  = tile >> 2;        // 0..23
      int seg = nn >> 3;          // 0=q 1=k 2=v
      int ntl = nn & 7;
      int e0  = seg * C_ + h * HD_ + ntl * 16;  // row into w_in
      f8v acc = {};
      for (int ks = 0; ks < 16; ++ks) {
        h16 a = ldfragA(Xs, LDX, mt * 16, ks * 32, lane);
        h16 b = ldfragB(w_in16, C_, e0, ks * 32, lane);
        acc = wmma_f16(a, b, acc);
      }
      float bias = b_in[e0 + (lane & 15)];
      int n = lane & 15;
      int t0 = mt * 16 + ((lane >> 4) << 3);
      int d = ntl * 16 + n;                // dim within head
      if (seg < 2) {                       // Q / K: row-major (t, d)
        _Float16* dst = (seg == 0) ? Qs : Ks;
#pragma unroll
        for (int i = 0; i < 8; ++i)
          dst[(t0 + i) * LDQ + d] = (_Float16)(acc[i] + bias);
      } else {                             // V: transposed (d, t), vector store
        h8 pk;
#pragma unroll
        for (int i = 0; i < 8; ++i) pk[i] = (_Float16)(acc[i] + bias);
        *(h8*)(Vt + (size_t)d * LDV + t0) = pk;
      }
    }
    __syncthreads();

    // ---- Stage 3: scores S = scale * Q @ K^T  (64x64, K-dim = 128) ----
    for (int tile = wave; tile < 16; tile += 8) {
      int mt = tile & 3, nt = tile >> 2;
      f8v acc = {};
      for (int ks = 0; ks < 4; ++ks) {
        h16 a = ldfragA(Qs, LDQ, mt * 16, ks * 32, lane);
        h16 b = ldfragB(Ks, LDQ, nt * 16, ks * 32, lane);
        acc = wmma_f16(a, b, acc);
      }
      int n = nt * 16 + (lane & 15);
      int t0 = mt * 16 + ((lane >> 4) << 3);
#pragma unroll
      for (int i = 0; i < 8; ++i)
        Sb[(t0 + i) * LDSS + n] = acc[i] * scale;
    }
    __syncthreads();

    // ---- Stage 4: row softmax (one thread per row) -> Pb f16 ----
    if (tid < L_) {
      float m = -3.0e38f;
      for (int k = 0; k < L_; ++k) m = fmaxf(m, Sb[tid * LDSS + k]);
      float sum = 0.0f;
      for (int k = 0; k < L_; ++k) {
        float e = __expf(Sb[tid * LDSS + k] - m);
        Sb[tid * LDSS + k] = e;
        sum += e;
      }
      float inv = 1.0f / sum;
      for (int k = 0; k < L_; ++k)
        Pb[tid * LDP + k] = (_Float16)(Sb[tid * LDSS + k] * inv);
    }
    __syncthreads();

    // ---- Stage 5: ctx_h = P @ V  (64x128, K-dim = 64) -> Cx[:, h*128 ..] ----
    for (int tile = wave; tile < 32; tile += 8) {
      int mt = tile & 3, nt = tile >> 2;    // nt 0..7
      f8v acc = {};
      for (int ks = 0; ks < 2; ++ks) {
        h16 a = ldfragA(Pb, LDP, mt * 16, ks * 32, lane);
        h16 b = ldfragB(Vt, LDV, nt * 16, ks * 32, lane);
        acc = wmma_f16(a, b, acc);
      }
      int d = nt * 16 + (lane & 15);
      int t0 = mt * 16 + ((lane >> 4) << 3);
#pragma unroll
      for (int i = 0; i < 8; ++i)
        Cx[(t0 + i) * LDX + h * HD_ + d] = (_Float16)acc[i];
    }
    __syncthreads();   // Qs/Ks/Vt/Sb/Pb reused by next head
  }

  // ---- Stage 6: out = ctx @ Wout^T + b_out, scattered window-merge store ----
  for (int tile = wave; tile < 128; tile += 8) {
    int mt = tile & 3, nt = tile >> 2;      // nt 0..31
    int o0 = nt * 16;
    f8v acc = {};
    for (int ks = 0; ks < 16; ++ks) {
      h16 a = ldfragA(Cx, LDX, mt * 16, ks * 32, lane);
      h16 b = ldfragB(w_out16, C_, o0, ks * 32, lane);
      acc = wmma_f16(a, b, acc);
    }
    int o = o0 + (lane & 15);
    float bias = b_out[o];
    int t0 = mt * 16 + ((lane >> 4) << 3);
#pragma unroll
    for (int i = 0; i < 8; ++i) {
      int t = t0 + i;
      int r = t >> 3, s = t & 7;
      out[(((size_t)bb * C_ + o) * HW_ + (row0 + r)) * HW_ + (col0 + s)] = acc[i] + bias;
    }
  }
}

// ---------------------------------------------------------------------------
extern "C" void kernel_launch(void* const* d_in, const int* in_sizes, int n_in,
                              void* d_out, int out_size, void* d_ws, size_t ws_size,
                              hipStream_t stream) {
  const float* x     = (const float*)d_in[0];
  const float* w_in  = (const float*)d_in[1];
  const float* b_in  = (const float*)d_in[2];
  const float* w_out = (const float*)d_in[3];
  const float* b_out = (const float*)d_in[4];
  float* out = (float*)d_out;

  _Float16* w16 = (_Float16*)d_ws;           // [0, 1536*512): w_in f16, then w_out f16
  const int n_in_w  = 3 * C_ * C_;           // 786432
  const int n_total = n_in_w + C_ * C_;      // + 262144

  cvt_weights_kernel<<<(n_total + 255) / 256, 256, 0, stream>>>(
      w_in, w_out, w16, n_in_w, n_total);

  win_attn_kernel<<<2048, 256, 0, stream>>>(
      x, w16, b_in, w16 + n_in_w, b_out, out);
}
